// MultiHeadRelativeAttention_31782757990578
// MI455X (gfx1250) — matrix operand order repaired and checked by reading, not compiled, not run on hardware
//
#include <hip/hip_runtime.h>
#include <hip/hip_fp16.h>

// ---------------------------------------------------------------------------
// MI455X (gfx1250) multi-head relative-position attention, wave32 + WMMA.
// B=2, T=2048, D_MODEL=1024, H=16, D_HEAD=64, NUM_BUCKETS=32, MAX_DIST=128.
// ---------------------------------------------------------------------------

#define BB   2
#define TT   2048
#define DM   1024
#define NH   16
#define DH   64
#define MROWS (BB*TT)          // 4096

typedef __attribute__((ext_vector_type(16))) _Float16 v16h;
typedef __attribute__((ext_vector_type(8)))  float    v8f;
typedef __attribute__((ext_vector_type(4)))  int      v4i;

// Async global->LDS (CDNA5): guarded so absence falls back to sync b128 copies.
#if defined(__gfx1250__) && \
    __has_builtin(__builtin_amdgcn_global_load_async_to_lds_b128) && \
    __has_builtin(__builtin_amdgcn_s_wait_asynccnt)
#define USE_ASYNC_LDS 1
#else
#define USE_ASYNC_LDS 0
#endif

union F16Frag { v16h v; uint4 u[2]; _Float16 h[16]; };
union F32Acc  { v8f  v; float f[8]; };

__device__ __forceinline__ v16h load_frag16(const _Float16* p0, const _Float16* p1) {
    F16Frag f;
    f.u[0] = *(const uint4*)p0;   // halves 0..7
    f.u[1] = *(const uint4*)p1;   // halves 8..15
    return f.v;
}

__device__ __forceinline__ v8f wmma16(v16h a, v16h b, v8f c) {
    // D = A(16x32 f16) x B(32x16 f16) + C(16x16 f32)
    return __builtin_amdgcn_wmma_f32_16x16x32_f16(
        false, a, false, b, (short)0, c, false, false);
}

#if USE_ASYNC_LDS
// Builtin signature (from toolchain diagnostic): (v4i* as1, v4i* as3, Ii, Ii)
typedef __attribute__((address_space(1))) v4i* gas_v4i;
typedef __attribute__((address_space(3))) v4i* las_v4i;
__device__ __forceinline__ void async_b128(const _Float16* g, _Float16* l) {
    __builtin_amdgcn_global_load_async_to_lds_b128(
        (gas_v4i)g, (las_v4i)l, 0, 0);
}
#endif

// T5 bidirectional relative-position bucket (num_buckets=32, max_dist=128).
__device__ __forceinline__ int rel_bucket(int rel) {
    int bucket = (rel > 0) ? 16 : 0;          // nb = 16
    int ad = rel < 0 ? -rel : rel;
    if (ad < 8) return bucket + ad;           // max_exact = 8
    int large = 8 + (int)(__logf((float)ad * 0.125f) * 2.8853901817f);
    return bucket + (large < 15 ? large : 15);
}

// ---------------------------------------------------------------------------
// Kernel 1/4: WMMA GEMM  C[M,N] = A[M,K] * Bw[K,N]
//   TA    : float (convert to f16 while staging) or _Float16 (direct/async)
//   BHTD  : true  -> f16 out in [B, H, T, DH] ; false -> TO row-major [M, N]
// 128 threads = 4 waves; block tile 64x64; K chunk 32.
// ---------------------------------------------------------------------------
template <typename TA, bool BHTD, typename TO>
__global__ __launch_bounds__(128) void gemm_wmma_kernel(
    const TA* __restrict__ A, const float* __restrict__ Bw,
    TO* __restrict__ out, int M, int N, int K)
{
    __shared__ __align__(16) _Float16 As[64][56];   // [row][k], 112B stride
    __shared__ __align__(16) _Float16 Bt[64][56];   // [n][k] transposed

    const int t    = threadIdx.x;
    const int w    = t >> 5;
    const int lane = t & 31;
    const int ln   = lane & 15;
    const int lh   = lane >> 4;
    const int m0   = blockIdx.x * 64;
    const int n0   = blockIdx.y * 64;

    // staging assignments (16 elements per thread per tile)
    const int ar = t >> 1, ac = (t & 1) * 16;   // A: 64 rows x 32 k
    const int br = t >> 2, bc = (t & 3) * 16;   // B: 32 k-rows x 64 cols

    const TA*    aBase = A  + (size_t)(m0 + ar) * K + ac;   // + kk per chunk
    const float* bBase = Bw + (size_t)br * N + n0 + bc;     // + kk*N per chunk

    F32Acc acc[4];
    #pragma unroll
    for (int nt = 0; nt < 4; nt++)
        #pragma unroll
        for (int j = 0; j < 8; j++) acc[nt].f[j] = 0.0f;

    for (int kk = 0; kk < K; kk += 32) {
        // ---- stage A tile ----
        const TA* ap = aBase + kk;
        if (sizeof(TA) == 2) {
#if USE_ASYNC_LDS
            async_b128((const _Float16*)ap,     &As[ar][ac]);
            async_b128((const _Float16*)ap + 8, &As[ar][ac + 8]);
#else
            *(uint4*)&As[ar][ac]     = ((const uint4*)ap)[0];
            *(uint4*)&As[ar][ac + 8] = ((const uint4*)ap)[1];
#endif
        } else {
            #pragma unroll
            for (int i = 0; i < 4; i++) {
                float4 x = ((const float4*)ap)[i];
                As[ar][ac + i * 4 + 0] = (_Float16)x.x;
                As[ar][ac + i * 4 + 1] = (_Float16)x.y;
                As[ar][ac + i * 4 + 2] = (_Float16)x.z;
                As[ar][ac + i * 4 + 3] = (_Float16)x.w;
            }
        }
        // ---- stage B tile (transpose to [n][k]) ----
        const float* bp = bBase + (size_t)kk * N;
        #pragma unroll
        for (int i = 0; i < 4; i++) {
            float4 x = ((const float4*)bp)[i];
            Bt[bc + i * 4 + 0][br] = (_Float16)x.x;
            Bt[bc + i * 4 + 1][br] = (_Float16)x.y;
            Bt[bc + i * 4 + 2][br] = (_Float16)x.z;
            Bt[bc + i * 4 + 3][br] = (_Float16)x.w;
        }
        if (kk + 32 < K) {
            __builtin_prefetch(ap + 32, 0, 1);
            __builtin_prefetch(bp + (size_t)32 * N, 0, 1);
        }
#if USE_ASYNC_LDS
        if (sizeof(TA) == 2) __builtin_amdgcn_s_wait_asynccnt(0);
#endif
        __syncthreads();

        const int mrow = w * 16 + ln;
        const int kb   = lh * 8;
        v16h af = load_frag16(&As[mrow][kb], &As[mrow][16 + kb]);
        const int kb2 = lh * 16;
        #pragma unroll
        for (int nt = 0; nt < 4; nt++) {
            int ncol = nt * 16 + ln;
            v16h bf = load_frag16(&Bt[ncol][kb], &Bt[ncol][16 + kb]);
            acc[nt].v = wmma16(af, bf, acc[nt].v);
        }
        __syncthreads();
    }

    // ---- store (block-uniform bases, int offsets) ----
    if (BHTD) {
        const int bidx = m0 >> 11;                         // batch (block-uniform)
        TO* obase = out + (((size_t)bidx * NH + blockIdx.y) * TT + (m0 & (TT - 1))) * DH;
        #pragma unroll
        for (int nt = 0; nt < 4; nt++)
            #pragma unroll
            for (int j = 0; j < 8; j++) {
                int ml = w * 16 + j + 8 * lh;
                obase[ml * DH + nt * 16 + ln] = (TO)acc[nt].f[j];
            }
    } else {
        TO* obase = out + (size_t)m0 * N + n0;
        #pragma unroll
        for (int nt = 0; nt < 4; nt++)
            #pragma unroll
            for (int j = 0; j < 8; j++) {
                int ml = w * 16 + j + 8 * lh;
                obase[ml * N + nt * 16 + ln] = (TO)acc[nt].f[j];
            }
    }
}

// ---------------------------------------------------------------------------
// Kernel 2/4: position bias [H, T, T] -> d_out second segment (268 MB stream).
// ---------------------------------------------------------------------------
__global__ __launch_bounds__(256) void bias_kernel(
    const float* __restrict__ rel_bias, float* __restrict__ out)
{
    size_t idx = (size_t)blockIdx.x * 256 + threadIdx.x;   // < 2^26
    int hh  = (int)(idx >> 22);
    int rem = (int)(idx & ((1u << 22) - 1));
    int q = rem >> 11;
    int k = rem & (TT - 1);
    out[idx] = rel_bias[rel_bucket(k - q) * NH + hh];
}

// ---------------------------------------------------------------------------
// Kernel 3/4: flash attention per (b, h, 64 q-rows).
// 4 waves, each owns a 16-row q tile and a 16x64 f32 O accumulator.
// K and V chunks (32 kv) staged row-major in LDS (async path when available).
// ---------------------------------------------------------------------------
__global__ __launch_bounds__(128) void attention_kernel(
    const _Float16* __restrict__ QH, const _Float16* __restrict__ KH,
    const _Float16* __restrict__ VH, const int* __restrict__ mask,
    const float* __restrict__ rel_bias, _Float16* __restrict__ O)
{
    __shared__ __align__(16) _Float16 Ks[32][72];      // [kv][d] 144B stride
    __shared__ __align__(16) _Float16 Vs[32][72];      // [kv][d] row-major
    __shared__ __align__(16) _Float16 Ps[4][16][32];   // per-wave P tile

    const int t    = threadIdx.x;
    const int w    = t >> 5;
    const int lane = t & 31;
    const int ln   = lane & 15;
    const int lh   = lane >> 4;
    const int hh   = blockIdx.y;
    const int b    = blockIdx.z;
    const int q0w  = blockIdx.x * 64 + w * 16;
    const size_t headBase = ((size_t)b * NH + hh) * TT * DH;

    const _Float16* kSrc  = KH + headBase;
    const _Float16* vSrc  = VH + headBase;
    const int*      mbase = mask + ((size_t)b * TT + q0w) * TT;

    // staging: 16 halves / thread / tile
    const int sr = t >> 2, sc = (t & 3) * 16;

    // Q A-fragments for d 0..31 and 32..63 (registers for all kv)
    v16h qA0, qA1;
    {
        const _Float16* qrow = QH + headBase + (size_t)q0w * DH;
        const int qo = ln * DH + lh * 8;
        qA0 = load_frag16(qrow + qo,      qrow + qo + 16);
        qA1 = load_frag16(qrow + qo + 32, qrow + qo + 48);
    }

    F32Acc o[4];
    float mval[8], lval[8];
    #pragma unroll
    for (int j = 0; j < 8; j++) {
        mval[j] = -1e30f; lval[j] = 0.0f;
        #pragma unroll
        for (int dt = 0; dt < 4; dt++) o[dt].f[j] = 0.0f;
    }

    for (int kv0 = 0; kv0 < TT; kv0 += 32) {
        // ---- stage K/V chunk [32][64] ----
        const int goff = (kv0 + sr) * DH + sc;
#if USE_ASYNC_LDS
        async_b128(kSrc + goff,     &Ks[sr][sc]);
        async_b128(kSrc + goff + 8, &Ks[sr][sc + 8]);
        async_b128(vSrc + goff,     &Vs[sr][sc]);
        async_b128(vSrc + goff + 8, &Vs[sr][sc + 8]);
#else
        *(uint4*)&Ks[sr][sc]     = *(const uint4*)(kSrc + goff);
        *(uint4*)&Ks[sr][sc + 8] = *(const uint4*)(kSrc + goff + 8);
        *(uint4*)&Vs[sr][sc]     = *(const uint4*)(vSrc + goff);
        *(uint4*)&Vs[sr][sc + 8] = *(const uint4*)(vSrc + goff + 8);
#endif
        if (kv0 + 32 < TT) {
            __builtin_prefetch(kSrc + goff + 32 * DH, 0, 1);
            __builtin_prefetch(vSrc + goff + 32 * DH, 0, 1);
        }
#if USE_ASYNC_LDS
        __builtin_amdgcn_s_wait_asynccnt(0);
#endif
        __syncthreads();

        // ---- scores: S[16 q][32 kv] = Q(16x64) * K^T(64x32) ----
        F32Acc s[2];
        #pragma unroll
        for (int sub = 0; sub < 2; sub++) {
            #pragma unroll
            for (int j = 0; j < 8; j++) s[sub].f[j] = 0.0f;
            const int kvl = sub * 16 + ln;
            const int kB  = lh * 8;
            v16h b0 = load_frag16(&Ks[kvl][kB],      &Ks[kvl][kB + 16]);       // d 0..31
            v16h b1 = load_frag16(&Ks[kvl][32 + kB], &Ks[kvl][32 + kB + 16]);  // d 32..63
            s[sub].v = wmma16(qA0, b0, s[sub].v);
            s[sub].v = wmma16(qA1, b1, s[sub].v);
        }

        // ---- scale + bias + mask, online softmax ----
        #pragma unroll
        for (int j = 0; j < 8; j++) {
            const int ml  = j + 8 * lh;          // local q row
            const int gq  = q0w + ml;
            const int gk0 = kv0 + ln;
            const int moff = ml * TT + kv0 + ln;
            float sc0 = s[0].f[j] * 0.125f + rel_bias[rel_bucket(gk0 - gq) * NH + hh];
            float sc1 = s[1].f[j] * 0.125f + rel_bias[rel_bucket(gk0 + 16 - gq) * NH + hh];
            if (mbase[moff]      == 0) sc0 = -1e9f;
            if (mbase[moff + 16] == 0) sc1 = -1e9f;

            float rm = fmaxf(sc0, sc1);
            rm = fmaxf(rm, __shfl_xor(rm, 1, 32));
            rm = fmaxf(rm, __shfl_xor(rm, 2, 32));
            rm = fmaxf(rm, __shfl_xor(rm, 4, 32));
            rm = fmaxf(rm, __shfl_xor(rm, 8, 32));

            float mnew = fmaxf(mval[j], rm);
            float corr = __expf(mval[j] - mnew);
            mval[j] = mnew;
            float p0 = __expf(sc0 - mnew);
            float p1 = __expf(sc1 - mnew);
            float rs = p0 + p1;
            rs += __shfl_xor(rs, 1, 32);
            rs += __shfl_xor(rs, 2, 32);
            rs += __shfl_xor(rs, 4, 32);
            rs += __shfl_xor(rs, 8, 32);
            lval[j] = lval[j] * corr + rs;
            #pragma unroll
            for (int dt = 0; dt < 4; dt++) o[dt].f[j] *= corr;

            Ps[w][ml][ln]      = (_Float16)p0;
            Ps[w][ml][16 + ln] = (_Float16)p1;
        }
        // wave-local DS RAW: same-wave LDS ops are in-order; compiler emits
        // the s_wait_dscnt before the fragment reloads below.

        // ---- O += P(16x32) * V(32x64) ----
        const int kb = lh * 16;
        v16h pA = load_frag16(&Ps[w][ln][kb], &Ps[w][ln][kb + 8]);
        #pragma unroll
        for (int dt = 0; dt < 4; dt++) {
            // B fragment of V: lane col = dt*16+ln, k = lh*16 + h (strided rows)
            F16Frag vf;
            const int dcol = dt * 16 + ln;
            #pragma unroll
            for (int h2 = 0; h2 < 16; h2++) vf.h[h2] = Vs[lh * 16 + h2][dcol];
            o[dt].v = wmma16(pA, vf.v, o[dt].v);
        }
        __syncthreads();
    }

    // ---- epilogue: O /= l, store f16 [B, T, D_MODEL] ----
    _Float16* obase = O + ((size_t)b * TT + q0w) * DM + hh * DH;
    #pragma unroll
    for (int j = 0; j < 8; j++) {
        const float inv = 1.0f / lval[j];
        const int ml = j + 8 * lh;
        #pragma unroll
        for (int dt = 0; dt < 4; dt++) {
            obase[ml * DM + dt * 16 + ln] = (_Float16)(o[dt].f[j] * inv);
        }
    }
}

// ---------------------------------------------------------------------------
extern "C" void kernel_launch(void* const* d_in, const int* in_sizes, int n_in,
                              void* d_out, int out_size, void* d_ws, size_t ws_size,
                              hipStream_t stream) {
    const float* q        = (const float*)d_in[0];
    const float* k        = (const float*)d_in[1];
    const float* v        = (const float*)d_in[2];
    const int*   mask     = (const int*)d_in[3];
    const float* Wq       = (const float*)d_in[4];
    const float* Wk       = (const float*)d_in[5];
    const float* Wv       = (const float*)d_in[6];
    const float* Wfc      = (const float*)d_in[7];
    const float* rel_bias = (const float*)d_in[8];

    float* out      = (float*)d_out;
    float* bias_out = out + (size_t)BB * TT * DM;       // second tuple output

    // workspace: 4 x (4096*1024) f16 buffers = 32 MB
    _Float16* QH = (_Float16*)d_ws;
    _Float16* KH = QH + (size_t)MROWS * DM;
    _Float16* VH = KH + (size_t)MROWS * DM;
    _Float16* OW = VH + (size_t)MROWS * DM;

    const dim3 gGemm(MROWS / 64, DM / 64, 1);   // 64 x 16
    const dim3 tGemm(128, 1, 1);

    // projections (f32 in -> f16 [B,H,T,DH] out)
    gemm_wmma_kernel<float, true, _Float16><<<gGemm, tGemm, 0, stream>>>(q, Wq, QH, MROWS, DM, DM);
    gemm_wmma_kernel<float, true, _Float16><<<gGemm, tGemm, 0, stream>>>(k, Wk, KH, MROWS, DM, DM);
    gemm_wmma_kernel<float, true, _Float16><<<gGemm, tGemm, 0, stream>>>(v, Wv, VH, MROWS, DM, DM);

    // position bias stream (independent of the GEMMs)
    bias_kernel<<<dim3((NH * TT * TT) / 256), dim3(256), 0, stream>>>(rel_bias, bias_out);

    // flash attention
    attention_kernel<<<dim3(TT / 64, NH, BB), dim3(128), 0, stream>>>(QH, KH, VH, mask, rel_bias, OW);

    // output projection (f16 in -> f32 row-major out)
    gemm_wmma_kernel<_Float16, false, float><<<gGemm, tGemm, 0, stream>>>(OW, Wfc, out, MROWS, DM, DM);
}
